// LanguageModelLSTM1_1477468750439
// MI455X (gfx1250) — matrix operand order repaired and checked by reading, not compiled, not run on hardware
//
#include <hip/hip_runtime.h>
#include <stdint.h>

// ---------------------------------------------------------------------------
// Problem constants (from reference)
// ---------------------------------------------------------------------------
#define CB   128     // batch
#define CT   20      // time steps
#define CVOC 12000   // vocab
#define CE   512     // embed dim
#define CH   512     // hidden
#define CVIS 512     // visual dim
#define CNV  64      // attention tokens

// ---------------------------------------------------------------------------
// Types / helpers
// ---------------------------------------------------------------------------
typedef __attribute__((ext_vector_type(16))) __bf16 v16bf;
typedef __attribute__((ext_vector_type(8)))  float  v8f;

struct U4 { unsigned int x, y, z, w; } __attribute__((aligned(16)));
union Frag { v16bf v; U4 q[2]; };

__device__ __forceinline__ unsigned short f2bf(float f) {
  union { float f; unsigned u; } x; x.f = f;
  unsigned r = x.u + 0x7FFFu + ((x.u >> 16) & 1u);   // round-to-nearest-even
  return (unsigned short)(r >> 16);
}
__device__ __forceinline__ float sigmoidf_(float x) { return 1.f / (1.f + expf(-x)); }

// ---------------------------------------------------------------------------
// Generic WMMA GEMM:  C[M,N] = A[M,K](bf16, row-major) x W[N,K](bf16,row-major)^T
// Block = 256 threads = 8 waves arranged 2(M) x 4(N).
// Wave tile: 32 rows x 32 cols -> 4 WMMA accs, A/B fragments each reused 2x
// (2 global_load_b128 per v_wmma).  Block tile: 64 x 128.
// K must be a multiple of 32; A/W rows 16B-aligned.
// EPI 0: C = acc + bias[n] (bias may be null)
// EPI 1: masked logits: row m -> (b = m/Tsteps, t = m%Tsteps);
//        C = (lengths[b] >= t) ? acc + bias[n] : 0
// ---------------------------------------------------------------------------
template <int EPI>
__global__ __launch_bounds__(256)
void wmma_gemm_nt(const unsigned short* __restrict__ A,
                  const unsigned short* __restrict__ W,
                  float* __restrict__ C,
                  const float* __restrict__ bias,
                  int M, int N, int K,
                  const int* __restrict__ lengths, int Tsteps)
{
  const int lane  = threadIdx.x & 31;
  const int wave  = threadIdx.x >> 5;
  const int half  = lane >> 4;       // 0..1
  const int l16   = lane & 15;       // 0..15
  const int waveM = wave & 1;        // 2 waves along M
  const int waveN = wave >> 1;       // 4 waves along N
  const int rowBase = blockIdx.y * 64 + waveM * 32;
  const int colBase = blockIdx.x * 128 + waveN * 32;

  // clamp load indices (stores are guarded, so clamping is value-safe)
  const int am0 = min(rowBase + l16,      M - 1);
  const int am1 = min(rowBase + 16 + l16, M - 1);
  const int bn0 = min(colBase + l16,      N - 1);
  const int bn1 = min(colBase + 16 + l16, N - 1);
  const unsigned short* Arow0 = A + (size_t)am0 * K;
  const unsigned short* Arow1 = A + (size_t)am1 * K;
  const unsigned short* Brow0 = W + (size_t)bn0 * K;
  const unsigned short* Brow1 = W + (size_t)bn1 * K;

  v8f acc00 = {0.f, 0.f, 0.f, 0.f, 0.f, 0.f, 0.f, 0.f};
  v8f acc01 = acc00, acc10 = acc00, acc11 = acc00;

  for (int k0 = 0; k0 < K; k0 += 32) {
    // prefetch ~4 k-iterations ahead (speculative; OOB prefetch is dropped)
    if (k0 + 128 < K) {
      __builtin_prefetch((const void*)(Arow0 + k0 + 128), 0, 3);
      __builtin_prefetch((const void*)(Brow0 + k0 + 128), 0, 3);
    }
    Frag a0, a1, b0, b1;
    // A 16x32 bf16 fragment: lane (half,m=l16): elems 0..7 -> K=k0+8*half+i,
    // elems 8..15 -> K=k0+16+8*half+(i-8)   (ISA 16-bit A layout)
    a0.q[0] = *(const U4*)(Arow0 + k0 + 8 * half);
    a0.q[1] = *(const U4*)(Arow0 + k0 + 16 + 8 * half);
    a1.q[0] = *(const U4*)(Arow1 + k0 + 8 * half);
    a1.q[1] = *(const U4*)(Arow1 + k0 + 16 + 8 * half);
    // B 32x16 bf16 fragment (col n = l16): elems 0..7 -> K = k0 + 8*half + i,
    // elems 8..15 -> K = k0 + 16 + 8*half + (i-8)   (same as A)
    const unsigned short* p0 = Brow0 + k0 + 8 * half;
    b0.q[0] = *(const U4*)(p0);
    b0.q[1] = *(const U4*)(p0 + 16);
    const unsigned short* p1 = Brow1 + k0 + 8 * half;
    b1.q[0] = *(const U4*)(p1);
    b1.q[1] = *(const U4*)(p1 + 16);

    acc00 = __builtin_amdgcn_wmma_f32_16x16x32_bf16(false, a0.v, false, b0.v,
                                                    (short)0, acc00, false, false);
    acc01 = __builtin_amdgcn_wmma_f32_16x16x32_bf16(false, a0.v, false, b1.v,
                                                    (short)0, acc01, false, false);
    acc10 = __builtin_amdgcn_wmma_f32_16x16x32_bf16(false, a1.v, false, b0.v,
                                                    (short)0, acc10, false, false);
    acc11 = __builtin_amdgcn_wmma_f32_16x16x32_bf16(false, a1.v, false, b1.v,
                                                    (short)0, acc11, false, false);
  }

  const int n0 = colBase + l16;
  const int n1 = colBase + 16 + l16;
#pragma unroll
  for (int r = 0; r < 8; ++r) {
    const int mA = rowBase + 8 * half + r;      // C/D layout: VGPR r -> M = r + 8*half
    const int mB = mA + 16;
    if (EPI == 0) {
      const float bi0 = (bias && n0 < N) ? bias[n0] : 0.f;
      const float bi1 = (bias && n1 < N) ? bias[n1] : 0.f;
      if (mA < M) {
        if (n0 < N) C[(size_t)mA * N + n0] = acc00[r] + bi0;
        if (n1 < N) C[(size_t)mA * N + n1] = acc01[r] + bi1;
      }
      if (mB < M) {
        if (n0 < N) C[(size_t)mB * N + n0] = acc10[r] + bi0;
        if (n1 < N) C[(size_t)mB * N + n1] = acc11[r] + bi1;
      }
    } else {
      if (mA < M) {
        const int b = mA / Tsteps, t = mA % Tsteps;
        const bool keep = lengths[b] >= t;
        if (n0 < N) C[(size_t)mA * N + n0] = keep ? (acc00[r] + bias[n0]) : 0.f;
        if (n1 < N) C[(size_t)mA * N + n1] = keep ? (acc01[r] + bias[n1]) : 0.f;
      }
      if (mB < M) {
        const int b = mB / Tsteps, t = mB % Tsteps;
        const bool keep = lengths[b] >= t;
        if (n0 < N) C[(size_t)mB * N + n0] = keep ? (acc10[r] + bias[n0]) : 0.f;
        if (n1 < N) C[(size_t)mB * N + n1] = keep ? (acc11[r] + bias[n1]) : 0.f;
      }
    }
  }
}

// ---------------------------------------------------------------------------
// Elementwise / prep kernels
// ---------------------------------------------------------------------------
__global__ void k_f2bf(const float* __restrict__ src, unsigned short* __restrict__ dst, int n) {
  int i = blockIdx.x * blockDim.x + threadIdx.x;
  if (i < n) dst[i] = f2bf(src[i]);
}

// features (B,512,1,7,7) -> f_bf (B,49,512)
__global__ void k_extract_f(const float* __restrict__ feats, unsigned short* __restrict__ f_bf) {
  int i = blockIdx.x * blockDim.x + threadIdx.x;
  const int n = CB * 49 * CVIS;
  if (i >= n) return;
  int c = i & (CVIS - 1);
  int p = (i / CVIS) % 49;
  int b = i / (CVIS * 49);
  f_bf[i] = f2bf(feats[((size_t)b * CVIS + c) * 49 + p]);
}

// we[b,t,:] = embed_w[captions[b,t],:]  (f32 + bf16)
__global__ void k_embed(const float* __restrict__ embed_w, const int* __restrict__ captions,
                        float* __restrict__ we_f, unsigned short* __restrict__ we_bf) {
  int i = blockIdx.x * blockDim.x + threadIdx.x;
  const int n = CB * CT * CE;
  if (i >= n) return;
  int c = i & (CE - 1);
  int bt = i / CE;
  int tok = captions[bt];
  float v = embed_w[(size_t)tok * CE + c];
  we_f[i] = v;
  we_bf[i] = f2bf(v);
}

// Build features_all (B,64,512): rows 0..48 = L2-normalized fv, 49..63 = eng.
// One block per (b,p) row; 256 threads, 2 elems each.
__global__ __launch_bounds__(256)
void k_build_fall(const float* __restrict__ fv, const float* __restrict__ eng,
                  float* __restrict__ fall_f, unsigned short* __restrict__ fall_bf) {
  int bp = blockIdx.x;
  int b = bp >> 6, p = bp & 63;
  const float* src = (p < 49) ? fv + ((size_t)b * 49 + p) * CVIS
                              : eng + ((size_t)b * 15 + (p - 49)) * CVIS;
  int tid = threadIdx.x, lane = tid & 31, wave = tid >> 5;
  float x0 = src[tid], x1 = src[tid + 256];
  __shared__ float red[8];
  __shared__ float s_inv;
  float ss = x0 * x0 + x1 * x1;
  for (int off = 16; off > 0; off >>= 1) ss += __shfl_down(ss, off, 32);
  if (lane == 0) red[wave] = ss;
  __syncthreads();
  if (tid == 0) {
    float tot = 0.f;
    for (int i = 0; i < 8; ++i) tot += red[i];
    s_inv = (p < 49) ? 1.f / fmaxf(sqrtf(tot), 1e-12f) : 1.f;
  }
  __syncthreads();
  float inv = s_inv;
  size_t o = ((size_t)b * 64 + p) * CVIS;
  float y0 = x0 * inv, y1 = x1 * inv;
  fall_f[o + tid] = y0;          fall_f[o + tid + 256] = y1;
  fall_bf[o + tid] = f2bf(y0);   fall_bf[o + tid + 256] = f2bf(y1);
}

// Fused attention: scores -> softmax over 64 -> context. One block per (b,t), t in 0..18.
__global__ __launch_bounds__(256)
void k_attention(const float* __restrict__ att_fea,   // (B,64,512)
                 const float* __restrict__ att_h,     // (B,20,512), rows 0..18 used
                 const float* __restrict__ att_bias,  // (64,)
                 const float* __restrict__ att_w,     // (512,)
                 const float* __restrict__ fall_f,    // (B,64,512)
                 float* __restrict__ ctx)             // (B,19,512)
{
  int bt = blockIdx.x;
  int b = bt / (CT - 1), t = bt % (CT - 1);
  int lane = threadIdx.x & 31, wave = threadIdx.x >> 5;
  __shared__ float s_score[CNV];
  __shared__ float s_alpha[CNV];
  const float* fea_b = att_fea + (size_t)b * CNV * CVIS;
  const float* h_bt  = att_h + ((size_t)b * CT + t) * CVIS;

  for (int n = wave; n < CNV; n += 8) {
    float bn = att_bias[n];
    const float* fea_n = fea_b + (size_t)n * CVIS;
    float s = 0.f;
    for (int v = lane; v < CVIS; v += 32) {
      float x = fea_n[v] + h_bt[v] + bn;
      s += fmaxf(x, 0.f) * att_w[v];
    }
    for (int off = 16; off > 0; off >>= 1) s += __shfl_down(s, off, 32);
    if (lane == 0) s_score[n] = s;
  }
  __syncthreads();
  if (threadIdx.x == 0) {
    float mx = -3.4e38f;
    for (int n = 0; n < CNV; ++n) mx = fmaxf(mx, s_score[n]);
    float sum = 0.f;
    for (int n = 0; n < CNV; ++n) { float e = expf(s_score[n] - mx); s_alpha[n] = e; sum += e; }
    float invs = 1.f / sum;
    for (int n = 0; n < CNV; ++n) s_alpha[n] *= invs;
  }
  __syncthreads();
  const float* fall_b = fall_f + (size_t)b * CNV * CVIS;
  float* ctx_bt = ctx + ((size_t)b * (CT - 1) + t) * CVIS;
  for (int v = threadIdx.x; v < CVIS; v += 256) {
    float s = 0.f;
    for (int n = 0; n < CNV; ++n) s += s_alpha[n] * fall_b[(size_t)n * CVIS + v];
    ctx_bt[v] = s;
  }
}

// w_cat (2048,1536) = [w_ih (2048,1024) | w_hh (2048,512)] in bf16
__global__ void k_wcat(const float* __restrict__ w_ih, const float* __restrict__ w_hh,
                       unsigned short* __restrict__ wcat) {
  int i = blockIdx.x * blockDim.x + threadIdx.x;
  const int n = (4 * CH) * (CE + CVIS + CH);
  if (i >= n) return;
  int k = i % (CE + CVIS + CH);
  int r = i / (CE + CVIS + CH);
  float v = (k < CE + CVIS) ? w_ih[(size_t)r * (CE + CVIS) + k]
                            : w_hh[(size_t)r * CH + (k - (CE + CVIS))];
  wcat[i] = f2bf(v);
}

__global__ void k_bsum(const float* __restrict__ b_ih, const float* __restrict__ b_hh,
                       float* __restrict__ bsum) {
  int i = blockIdx.x * blockDim.x + threadIdx.x;
  if (i < 4 * CH) bsum[i] = b_ih[i] + b_hh[i];
}

__global__ void k_zero2(float* __restrict__ h, float* __restrict__ c, int n) {
  int i = blockIdx.x * blockDim.x + threadIdx.x;
  if (i < n) { h[i] = 0.f; c[i] = 0.f; }
}

// x_bf[b, 0:512]=feas_t, [512:1024]=we[b,t], [1024:1536]=h[b]
__global__ void k_build_x(const float* __restrict__ feas0, const float* __restrict__ ctx,
                          const float* __restrict__ we_f, const float* __restrict__ h,
                          unsigned short* __restrict__ x_bf, int t) {
  int i = blockIdx.x * blockDim.x + threadIdx.x;
  const int n = CB * (CVIS + CE + CH);
  if (i >= n) return;
  int c = i % (CVIS + CE + CH);
  int b = i / (CVIS + CE + CH);
  float v;
  if (c < CVIS) {
    v = (t == 0) ? feas0[(size_t)b * CVIS + c]
                 : ctx[((size_t)b * (CT - 1) + (t - 1)) * CVIS + c];
  } else if (c < CVIS + CE) {
    v = we_f[((size_t)b * CT + t) * CE + (c - CVIS)];
  } else {
    v = h[(size_t)b * CH + (c - CVIS - CE)];
  }
  x_bf[i] = f2bf(v);
}

// LSTM cell: gates (B,2048) f32 -> h,c (B,512); hs_bf[(b*T+t)*512+n] = bf16(h)
__global__ void k_lstm_cell(const float* __restrict__ gates, float* __restrict__ h,
                            float* __restrict__ c, unsigned short* __restrict__ hs_bf, int t) {
  int i = blockIdx.x * blockDim.x + threadIdx.x;
  if (i >= CB * CH) return;
  int n = i & (CH - 1);
  int b = i / CH;
  const float* g = gates + (size_t)b * 4 * CH;
  float ig = g[n], fg = g[CH + n], gg = g[2 * CH + n], og = g[3 * CH + n];
  float cc = sigmoidf_(fg) * c[i] + sigmoidf_(ig) * tanhf(gg);
  float hh = sigmoidf_(og) * tanhf(cc);
  c[i] = cc;
  h[i] = hh;
  hs_bf[((size_t)b * CT + t) * CH + n] = f2bf(hh);
}

// ---------------------------------------------------------------------------
// Host launcher
// ---------------------------------------------------------------------------
extern "C" void kernel_launch(void* const* d_in, const int* in_sizes, int n_in,
                              void* d_out, int out_size, void* d_ws, size_t ws_size,
                              hipStream_t stream) {
  const float* feature_0 = (const float*)d_in[0];
  const float* features  = (const float*)d_in[1];
  const float* eng       = (const float*)d_in[2];
  const int*   captions  = (const int*)d_in[3];
  const int*   lengths   = (const int*)d_in[4];
  const float* embed_w   = (const float*)d_in[5];
  const float* w_ih      = (const float*)d_in[6];
  const float* w_hh      = (const float*)d_in[7];
  const float* b_ih      = (const float*)d_in[8];
  const float* b_hh      = (const float*)d_in[9];
  const float* fc_w      = (const float*)d_in[10];
  const float* fc_b      = (const float*)d_in[11];
  const float* att_vw_w  = (const float*)d_in[12];
  const float* att_hw_w  = (const float*)d_in[13];
  const float* att_bias  = (const float*)d_in[14];
  const float* att_w_w   = (const float*)d_in[15];
  const float* att_fc_w  = (const float*)d_in[16];
  const float* att_fc_b  = (const float*)d_in[17];
  const float* img_w     = (const float*)d_in[18];
  const float* img_b     = (const float*)d_in[19];
  float* out = (float*)d_out;

  // workspace carve-up (all 256B aligned)
  char* p = (char*)d_ws;
  auto take = [&](size_t bytes) -> char* {
    char* r = p;
    p += (bytes + 255) & ~(size_t)255;
    return r;
  };
  unsigned short* bf_att_fc = (unsigned short*)take((size_t)CVIS * CVIS * 2);
  unsigned short* bf_att_vw = (unsigned short*)take((size_t)CVIS * CVIS * 2);
  unsigned short* bf_att_hw = (unsigned short*)take((size_t)CVIS * CE * 2);
  unsigned short* bf_img_w  = (unsigned short*)take((size_t)CVIS * 2048 * 2);
  unsigned short* bf_fc_w   = (unsigned short*)take((size_t)CVOC * CH * 2);
  unsigned short* bf_wcat   = (unsigned short*)take((size_t)(4 * CH) * (CVIS + CE + CH) * 2);
  unsigned short* bf_feat0  = (unsigned short*)take((size_t)CB * 2048 * 2);
  unsigned short* bf_f      = (unsigned short*)take((size_t)CB * 49 * CVIS * 2);
  float*          f32_fv    = (float*)take((size_t)CB * 49 * CVIS * 4);
  float*          f32_fall  = (float*)take((size_t)CB * CNV * CVIS * 4);
  unsigned short* bf_fall   = (unsigned short*)take((size_t)CB * CNV * CVIS * 2);
  float*          f32_we    = (float*)take((size_t)CB * CT * CE * 4);
  unsigned short* bf_we     = (unsigned short*)take((size_t)CB * CT * CE * 2);
  float*          f32_afea  = (float*)take((size_t)CB * CNV * CVIS * 4);
  float*          f32_ath   = (float*)take((size_t)CB * CT * CVIS * 4);
  float*          f32_ctx   = (float*)take((size_t)CB * (CT - 1) * CVIS * 4);
  float*          f32_feas0 = (float*)take((size_t)CB * CVIS * 4);
  float*          f32_bsum  = (float*)take((size_t)4 * CH * 4);
  unsigned short* bf_x      = (unsigned short*)take((size_t)CB * (CVIS + CE + CH) * 2);
  float*          f32_gates = (float*)take((size_t)CB * 4 * CH * 4);
  float*          f32_h     = (float*)take((size_t)CB * CH * 4);
  float*          f32_c     = (float*)take((size_t)CB * CH * 4);
  unsigned short* bf_hs     = (unsigned short*)take((size_t)CB * CT * CH * 2);

  auto cdiv = [](int a, int b) { return (a + b - 1) / b; };

  // --- weight conversions ---
  {
    int n;
    n = CVIS * CVIS;  k_f2bf<<<cdiv(n, 256), 256, 0, stream>>>(att_fc_w, bf_att_fc, n);
    n = CVIS * CVIS;  k_f2bf<<<cdiv(n, 256), 256, 0, stream>>>(att_vw_w, bf_att_vw, n);
    n = CVIS * CE;    k_f2bf<<<cdiv(n, 256), 256, 0, stream>>>(att_hw_w, bf_att_hw, n);
    n = CVIS * 2048;  k_f2bf<<<cdiv(n, 256), 256, 0, stream>>>(img_w, bf_img_w, n);
    n = CVOC * CH;    k_f2bf<<<cdiv(n, 256), 256, 0, stream>>>(fc_w, bf_fc_w, n);
    n = CB * 2048;    k_f2bf<<<cdiv(n, 256), 256, 0, stream>>>(feature_0, bf_feat0, n);
    n = (4 * CH) * (CVIS + CE + CH);
    k_wcat<<<cdiv(n, 256), 256, 0, stream>>>(w_ih, w_hh, bf_wcat);
    k_bsum<<<cdiv(4 * CH, 256), 256, 0, stream>>>(b_ih, b_hh, f32_bsum);
  }

  // --- input prep ---
  {
    int n = CB * 49 * CVIS;
    k_extract_f<<<cdiv(n, 256), 256, 0, stream>>>(features, bf_f);
    n = CB * CT * CE;
    k_embed<<<cdiv(n, 256), 256, 0, stream>>>(embed_w, captions, f32_we, bf_we);
  }

  // --- G1: fv = f @ att_fc_w^T + b  (M=B*49, N=512, K=512) ---
  wmma_gemm_nt<0><<<dim3(512 / 128, (CB * 49) / 64), 256, 0, stream>>>(
      bf_f, bf_att_fc, f32_fv, att_fc_b, CB * 49, CVIS, CVIS, nullptr, 0);

  // --- normalize + concat eng -> features_all (f32 + bf16) ---
  k_build_fall<<<CB * CNV, 256, 0, stream>>>(f32_fv, eng, f32_fall, bf_fall);

  // --- G3: att_fea = features_all @ att_vw_w^T  (M=B*64, N=512, K=512) ---
  wmma_gemm_nt<0><<<dim3(512 / 128, (CB * CNV) / 64), 256, 0, stream>>>(
      bf_fall, bf_att_vw, f32_afea, nullptr, CB * CNV, CVIS, CVIS, nullptr, 0);

  // --- G4: att_h = we @ att_hw_w^T  (M=B*20, N=512, K=512; row 19 unused) ---
  wmma_gemm_nt<0><<<dim3(512 / 128, (CB * CT) / 64), 256, 0, stream>>>(
      bf_we, bf_att_hw, f32_ath, nullptr, CB * CT, CVIS, CE, nullptr, 0);

  // --- G2: feas0 = feature_0 @ img_w^T + img_b  (M=128, N=512, K=2048) ---
  wmma_gemm_nt<0><<<dim3(512 / 128, CB / 64), 256, 0, stream>>>(
      bf_feat0, bf_img_w, f32_feas0, img_b, CB, CVIS, 2048, nullptr, 0);

  // --- fused attention: scores + softmax + context ---
  k_attention<<<CB * (CT - 1), 256, 0, stream>>>(f32_afea, f32_ath, att_bias, att_w_w,
                                                 f32_fall, f32_ctx);

  // --- LSTM ---
  k_zero2<<<cdiv(CB * CH, 256), 256, 0, stream>>>(f32_h, f32_c, CB * CH);
  for (int t = 0; t < CT; ++t) {
    int nx = CB * (CVIS + CE + CH);
    k_build_x<<<cdiv(nx, 256), 256, 0, stream>>>(f32_feas0, f32_ctx, f32_we, f32_h, bf_x, t);
    // gates = x @ [w_ih|w_hh]^T + (b_ih+b_hh)  (M=128, N=2048, K=1536)
    wmma_gemm_nt<0><<<dim3(4 * CH / 128, CB / 64), 256, 0, stream>>>(
        bf_x, bf_wcat, f32_gates, f32_bsum, CB, 4 * CH, CVIS + CE + CH, nullptr, 0);
    k_lstm_cell<<<cdiv(CB * CH, 256), 256, 0, stream>>>(f32_gates, f32_h, f32_c, bf_hs, t);
  }

  // --- G6: logits = hs @ fc_w^T + fc_b, masked, straight into d_out ---
  wmma_gemm_nt<1><<<dim3((CVOC + 127) / 128, (CB * CT) / 64), 256, 0, stream>>>(
      bf_hs, bf_fc_w, out, fc_b, CB * CT, CVOC, CH, lengths, CT);

  (void)in_sizes; (void)n_in; (void)out_size; (void)ws_size;
}
